// LongRec_18202071400579
// MI455X (gfx1250) — matrix operand order repaired and checked by reading, not compiled, not run on hardware
//
#include <hip/hip_runtime.h>
#include <hip/hip_bf16.h>
#include <math.h>

// ---------------------------------------------------------------------------
// LongRec GRU-variant, CDNA5 (gfx1250).
// Matrix math: v_wmma_f32_16x16x32_f16 (f16 A/B, fp32 accumulate).
// Data movement: GLOBAL_LOAD_ASYNC_TO_LDS (ASYNCcnt) double-buffering where
// the toolchain exposes the gfx1250 builtins, synchronous fallback otherwise.
// Attention branch removed: softmax over a singleton axis == 1 -> h_att = h_d.
// ---------------------------------------------------------------------------

#define BB   256      // batch
#define SS   200      // seq len
#define DD   128      // emb dim
#define UU   128      // gru units
#define CC   50000    // classes
#define MM   (BB*SS)  // flattened rows

typedef __attribute__((ext_vector_type(16))) _Float16 v16h;
typedef __attribute__((ext_vector_type(8)))  _Float16 v8h;
typedef __attribute__((ext_vector_type(8)))  float    v8f;
typedef __attribute__((ext_vector_type(4)))  float    v4f;
typedef __attribute__((ext_vector_type(4)))  int      v4i;

#if __has_builtin(__builtin_amdgcn_global_load_async_to_lds_b128) && \
    __has_builtin(__builtin_amdgcn_s_wait_asynccnt)
#define HAVE_ASYNC 1
// builtin signature: (int4 addrspace(1)*, int4 addrspace(3)*, imm, imm)
#define AS_G(p) ((__attribute__((address_space(1))) v4i*)                     \
                 (__attribute__((address_space(1))) void*)(void*)(p))
#define AS_L(p) ((__attribute__((address_space(3))) v4i*)                     \
                 (__attribute__((address_space(3))) void*)(void*)(p))
#else
#define HAVE_ASYNC 0
#endif

// ---------------- WMMA fragment helpers (wave32) ----------------------------

__device__ __forceinline__ v16h frag_cat(v8h lo, v8h hi) {
  v16h r;
#pragma unroll
  for (int i = 0; i < 8; ++i) { r[i] = lo[i]; r[i + 8] = hi[i]; }
  return r;
}

// A operand: 16xK row-major f16 source, leading dim ldk (halves), K-step k0.
// ISA layout: lanes 0-15 row=lane K={k0..k0+7, k0+16..k0+23};
//             lanes 16-31 row=lane-16 K={k0+8..k0+15, k0+24..k0+31}.
__device__ __forceinline__ v16h load_a_frag(const _Float16* base, int ldk,
                                            int k0, int lane) {
  const int row = lane & 15;
  const int kb  = k0 + ((lane < 16) ? 0 : 8);
  const _Float16* p = base + row * ldk + kb;
  v8h lo = *(const v8h*)(p);
  v8h hi = *(const v8h*)(p + 16);
  return frag_cat(lo, hi);
}

// B operand from transposed weights Wt[N][K] (f16 row-major).
// ISA layout: lanes 0-15 col=n K={k0..k0+7, k0+16..k0+23};
//             lanes 16-31 col=n K={k0+8..k0+15, k0+24..k0+31}.
__device__ __forceinline__ v16h load_b_frag(const _Float16* wt, int K,
                                            int ncol0, int k0, int lane) {
  const int n  = ncol0 + (lane & 15);
  const int kb = k0 + ((lane < 16) ? 0 : 8);
  const _Float16* p = wt + (size_t)n * (size_t)K + kb;
  v8h lo = *(const v8h*)(p);
  v8h hi = *(const v8h*)(p + 16);
  return frag_cat(lo, hi);
}

__device__ __forceinline__ v8f wmma_step(v16h a, v16h b, v8f c) {
  return __builtin_amdgcn_wmma_f32_16x16x32_f16(false, a, false, b,
                                                (short)0, c, false, false);
}

__device__ __forceinline__ float sigmoidf(float x) {
  return 1.0f / (1.0f + __expf(-x));
}
__device__ __forceinline__ float softplusf(float x) {
  return (x > 20.0f) ? x : log1pf(__expf(x));
}

// ---------------- Kernel 0: weight transpose + f16 convert ------------------
// dst[n*K + k] = (f16) src[k*N + n]
__global__ void wtrans_kernel(const float* __restrict__ src,
                              _Float16* __restrict__ dst, int K, int N) {
  size_t total = (size_t)K * (size_t)N;
  for (size_t i = (size_t)blockIdx.x * blockDim.x + threadIdx.x; i < total;
       i += (size_t)gridDim.x * blockDim.x) {
    size_t k = i / (size_t)N;
    size_t n = i % (size_t)N;
    dst[n * (size_t)K + k] = (_Float16)src[i];
  }
}

// ---------------- Kernel 1: fused gather + projections ----------------------
// One block = 16 rows of flattened (b,s); 8 waves, each owns 16 output cols.
// Embedding gather uses per-lane async global->LDS copies when available.
__global__ __launch_bounds__(256) void proj_kernel(
    const int* __restrict__ item, const int* __restrict__ tix,
    const int* __restrict__ frq,
    const float* __restrict__ emb_i, const float* __restrict__ emb_t,
    const float* __restrict__ emb_f,
    const _Float16* __restrict__ WtXr, const _Float16* __restrict__ WtXz,
    const _Float16* __restrict__ WtXh,
    const _Float16* __restrict__ WtXtg, const _Float16* __restrict__ WtTg,
    const _Float16* __restrict__ WtXfg, const _Float16* __restrict__ WtFg,
    const _Float16* __restrict__ WtDelta,
    const _Float16* __restrict__ WtFdir, const _Float16* __restrict__ WtPsi,
    const float* __restrict__ b_r, const float* __restrict__ b_z,
    const float* __restrict__ b_h,
    const float* __restrict__ b_tg, const float* __restrict__ b_fg,
    const float* __restrict__ b_delta,
    const float* __restrict__ b_fdir, const float* __restrict__ b_psi,
    _Float16* __restrict__ g_xr, _Float16* __restrict__ g_xz,
    _Float16* __restrict__ g_xh,
    _Float16* __restrict__ g_decay, _Float16* __restrict__ g_fdir,
    _Float16* __restrict__ g_psi) {
  __shared__ __align__(16) float    sRaw[3][16][DD];   // 24 KB fp32 staging
  __shared__ __align__(16) _Float16 sXi[16][DD];
  __shared__ __align__(16) _Float16 sXt[16][DD];
  __shared__ __align__(16) _Float16 sXf[16][DD];
  __shared__ __align__(16) _Float16 sTg[16][UU];
  __shared__ __align__(16) _Float16 sFg[16][UU];

  const int tid = threadIdx.x;
  const int m0  = blockIdx.x * 16;

  // --- async gather of 3 embedding rows per tile row into LDS (fp32) ---
  {
    const int row = tid >> 4;
    const int c0  = (tid & 15) * 8;          // 8 floats = 32 B = 2 x b128
    const int m   = m0 + row;
    const float* si = emb_i + (size_t)item[m] * DD + c0;
    const float* st = emb_t + (size_t)tix[m]  * DD + c0;
    const float* sf = emb_f + (size_t)frq[m]  * DD + c0;
#if HAVE_ASYNC
    __builtin_amdgcn_global_load_async_to_lds_b128(AS_G(si), AS_L(&sRaw[0][row][c0]), 0, 0);
    __builtin_amdgcn_global_load_async_to_lds_b128(AS_G(si), AS_L(&sRaw[0][row][c0]), 16, 0);
    __builtin_amdgcn_global_load_async_to_lds_b128(AS_G(st), AS_L(&sRaw[1][row][c0]), 0, 0);
    __builtin_amdgcn_global_load_async_to_lds_b128(AS_G(st), AS_L(&sRaw[1][row][c0]), 16, 0);
    __builtin_amdgcn_global_load_async_to_lds_b128(AS_G(sf), AS_L(&sRaw[2][row][c0]), 0, 0);
    __builtin_amdgcn_global_load_async_to_lds_b128(AS_G(sf), AS_L(&sRaw[2][row][c0]), 16, 0);
    __builtin_amdgcn_s_wait_asynccnt(0);
#else
    *(v4f*)&sRaw[0][row][c0]     = *(const v4f*)(si);
    *(v4f*)&sRaw[0][row][c0 + 4] = *(const v4f*)(si + 4);
    *(v4f*)&sRaw[1][row][c0]     = *(const v4f*)(st);
    *(v4f*)&sRaw[1][row][c0 + 4] = *(const v4f*)(st + 4);
    *(v4f*)&sRaw[2][row][c0]     = *(const v4f*)(sf);
    *(v4f*)&sRaw[2][row][c0 + 4] = *(const v4f*)(sf + 4);
#endif
  }
  __syncthreads();
  // --- convert staged fp32 -> f16 A-operand buffers ---
  for (int e = tid; e < 3 * 16 * DD; e += 256) {
    const int q  = e / (16 * DD);
    const int rr = (e / DD) & 15;
    const int c  = e & (DD - 1);
    const _Float16 v = (_Float16)sRaw[q][rr][c];
    if (q == 0)      sXi[rr][c] = v;
    else if (q == 1) sXt[rr][c] = v;
    else             sXf[rr][c] = v;
  }
  __syncthreads();

  const int lane  = tid & 31;
  const int w     = tid >> 5;
  const int ncol0 = w * 16;
  const int nloc  = lane & 15;
  const int col   = ncol0 + nloc;
  const int hi8   = (lane < 16) ? 0 : 8;

  // ---- tg = sigmoid(xi@Wxtg + xt@Wtg + b_tg) -> LDS (f16) ----
  {
    v8f acc; const float bb = b_tg[col];
#pragma unroll
    for (int i = 0; i < 8; ++i) acc[i] = bb;
#pragma unroll
    for (int k4 = 0; k4 < 4; ++k4)
      acc = wmma_step(load_a_frag(&sXi[0][0], DD, 32 * k4, lane),
                      load_b_frag(WtXtg, DD, ncol0, 32 * k4, lane), acc);
#pragma unroll
    for (int k4 = 0; k4 < 4; ++k4)
      acc = wmma_step(load_a_frag(&sXt[0][0], DD, 32 * k4, lane),
                      load_b_frag(WtTg, DD, ncol0, 32 * k4, lane), acc);
#pragma unroll
    for (int i = 0; i < 8; ++i) sTg[i + hi8][col] = (_Float16)sigmoidf(acc[i]);
  }
  // ---- fg = sigmoid(xi@Wxfg + xf@Wfg + b_fg) -> LDS (f16) ----
  {
    v8f acc; const float bb = b_fg[col];
#pragma unroll
    for (int i = 0; i < 8; ++i) acc[i] = bb;
#pragma unroll
    for (int k4 = 0; k4 < 4; ++k4)
      acc = wmma_step(load_a_frag(&sXi[0][0], DD, 32 * k4, lane),
                      load_b_frag(WtXfg, DD, ncol0, 32 * k4, lane), acc);
#pragma unroll
    for (int k4 = 0; k4 < 4; ++k4)
      acc = wmma_step(load_a_frag(&sXf[0][0], DD, 32 * k4, lane),
                      load_b_frag(WtFg, DD, ncol0, 32 * k4, lane), acc);
#pragma unroll
    for (int i = 0; i < 8; ++i) sFg[i + hi8][col] = (_Float16)sigmoidf(acc[i]);
  }

  // ---- xr / xz / xh = xi@W + b -> global f16 ----
  const _Float16* wx[3] = {WtXr, WtXz, WtXh};
  const float*    bx[3] = {b_r, b_z, b_h};
  _Float16*       gx[3] = {g_xr, g_xz, g_xh};
#pragma unroll
  for (int q = 0; q < 3; ++q) {
    v8f acc; const float bb = bx[q][col];
#pragma unroll
    for (int i = 0; i < 8; ++i) acc[i] = bb;
#pragma unroll
    for (int k4 = 0; k4 < 4; ++k4)
      acc = wmma_step(load_a_frag(&sXi[0][0], DD, 32 * k4, lane),
                      load_b_frag(wx[q], DD, ncol0, 32 * k4, lane), acc);
#pragma unroll
    for (int i = 0; i < 8; ++i)
      gx[q][(size_t)(m0 + i + hi8) * UU + col] = (_Float16)acc[i];
  }

  __syncthreads();  // sTg/sFg complete across all waves

  // ---- decay = exp(-softplus(concat(tg,fg)@W_delta + b_delta)) ----
  {
    v8f acc; const float bb = b_delta[col];
#pragma unroll
    for (int i = 0; i < 8; ++i) acc[i] = bb;
#pragma unroll
    for (int k4 = 0; k4 < 4; ++k4)
      acc = wmma_step(load_a_frag(&sTg[0][0], UU, 32 * k4, lane),
                      load_b_frag(WtDelta, 2 * UU, ncol0, 32 * k4, lane), acc);
#pragma unroll
    for (int k4 = 0; k4 < 4; ++k4)
      acc = wmma_step(load_a_frag(&sFg[0][0], UU, 32 * k4, lane),
                      load_b_frag(WtDelta, 2 * UU, ncol0, UU + 32 * k4, lane),
                      acc);
#pragma unroll
    for (int i = 0; i < 8; ++i)
      g_decay[(size_t)(m0 + i + hi8) * UU + col] =
          (_Float16)__expf(-softplusf(acc[i]));
  }
  // ---- fdir = fg@W_f_dir + b ----
  {
    v8f acc; const float bb = b_fdir[col];
#pragma unroll
    for (int i = 0; i < 8; ++i) acc[i] = bb;
#pragma unroll
    for (int k4 = 0; k4 < 4; ++k4)
      acc = wmma_step(load_a_frag(&sFg[0][0], UU, 32 * k4, lane),
                      load_b_frag(WtFdir, UU, ncol0, 32 * k4, lane), acc);
#pragma unroll
    for (int i = 0; i < 8; ++i)
      g_fdir[(size_t)(m0 + i + hi8) * UU + col] = (_Float16)acc[i];
  }
  // ---- psi = sigmoid(fg@W_psi + b) ----
  {
    v8f acc; const float bb = b_psi[col];
#pragma unroll
    for (int i = 0; i < 8; ++i) acc[i] = bb;
#pragma unroll
    for (int k4 = 0; k4 < 4; ++k4)
      acc = wmma_step(load_a_frag(&sFg[0][0], UU, 32 * k4, lane),
                      load_b_frag(WtPsi, UU, ncol0, 32 * k4, lane), acc);
#pragma unroll
    for (int i = 0; i < 8; ++i)
      g_psi[(size_t)(m0 + i + hi8) * UU + col] = (_Float16)sigmoidf(acc[i]);
  }
}

// ---------------- Kernel 2: sequential recurrent scan -----------------------
// 16 blocks; each owns 16 batch rows, 8 waves each own 16 hidden columns.
// W_hr/W_hz/W_hh column slices are pinned in registers (12 x v16h per lane).
// Per-step operand tensors are double-buffered into LDS with async copies.

// stage one step's 6 tensor slices [16 rows][128 f16] into LDS at dst
__device__ __forceinline__ void stage_step(const _Float16* t0,
                                           const _Float16* t1,
                                           const _Float16* t2,
                                           const _Float16* t3,
                                           const _Float16* t4,
                                           const _Float16* t5,
                                           int rb0, int t, _Float16* dst,
                                           int tid) {
  const int row = tid >> 4;
  const int c0  = (tid & 15) * 8;  // 8 halves = 16 B = one b128
  const size_t gofs = ((size_t)(rb0 + row) * SS + t) * UU + c0;
  const _Float16* srcs[6] = {t0 + gofs, t1 + gofs, t2 + gofs,
                             t3 + gofs, t4 + gofs, t5 + gofs};
#pragma unroll
  for (int q = 0; q < 6; ++q) {
    _Float16* d = dst + (q * 16 + row) * UU + c0;
#if HAVE_ASYNC
    __builtin_amdgcn_global_load_async_to_lds_b128(AS_G(srcs[q]), AS_L(d), 0, 0);
#else
    *(v8h*)d = *(const v8h*)srcs[q];
#endif
  }
}

__global__ __launch_bounds__(256) void scan_kernel(
    const _Float16* __restrict__ WtHr, const _Float16* __restrict__ WtHz,
    const _Float16* __restrict__ WtHh,
    const _Float16* __restrict__ g_xr, const _Float16* __restrict__ g_xz,
    const _Float16* __restrict__ g_xh,
    const _Float16* __restrict__ g_decay, const _Float16* __restrict__ g_fdir,
    const _Float16* __restrict__ g_psi,
    _Float16* __restrict__ g_h16) {
  // tensor order in stage: 0=decay 1=xr 2=xz 3=xh 4=fdir 5=psi
  __shared__ __align__(16) _Float16 sT[2][6][16][UU];  // 48 KB
  __shared__ __align__(16) _Float16 sHd[16][UU];       // 4 KB
  __shared__ __align__(16) _Float16 sRh[16][UU];       // 4 KB

  const int tid   = threadIdx.x;
  const int lane  = tid & 31;
  const int w     = tid >> 5;
  const int ncol0 = w * 16;
  const int nloc  = lane & 15;
  const int col   = ncol0 + nloc;
  const int hi8   = (lane < 16) ? 0 : 8;
  const int rb0   = blockIdx.x * 16;

  // Pin recurrent weight B-fragments in registers for the whole scan.
  v16h whr[4], whz[4], whh[4];
#pragma unroll
  for (int k4 = 0; k4 < 4; ++k4) {
    whr[k4] = load_b_frag(WtHr, UU, ncol0, 32 * k4, lane);
    whz[k4] = load_b_frag(WtHz, UU, ncol0, 32 * k4, lane);
    whh[k4] = load_b_frag(WtHh, UU, ncol0, 32 * k4, lane);
  }

  v8f h;
#pragma unroll
  for (int i = 0; i < 8; ++i) h[i] = 0.0f;

  int buf = 0;
  stage_step(g_decay, g_xr, g_xz, g_xh, g_fdir, g_psi, rb0, 0,
             &sT[0][0][0][0], tid);

  for (int t = 0; t < SS; ++t) {
    // issue next step's staging into the other buffer (overlaps this step)
    if (t + 1 < SS)
      stage_step(g_decay, g_xr, g_xz, g_xh, g_fdir, g_psi, rb0, t + 1,
                 &sT[buf ^ 1][0][0][0], tid);
#if HAVE_ASYNC
    if (t + 1 < SS) __builtin_amdgcn_s_wait_asynccnt(6);  // batch t landed
    else            __builtin_amdgcn_s_wait_asynccnt(0);
#endif
    __syncthreads();  // batch t visible to all waves

    const _Float16 (*st)[16][UU] = sT[buf];

    // h_d = decay * h ; publish to LDS as f16 A operand
    v8f hd;
#pragma unroll
    for (int i = 0; i < 8; ++i) {
      hd[i] = (float)st[0][i + hi8][col] * h[i];
      sHd[i + hi8][col] = (_Float16)hd[i];
    }
    __syncthreads();

    // r = sigmoid(xr + h_d @ W_hr)
    v8f r;
#pragma unroll
    for (int i = 0; i < 8; ++i) r[i] = (float)st[1][i + hi8][col];
#pragma unroll
    for (int k4 = 0; k4 < 4; ++k4)
      r = wmma_step(load_a_frag(&sHd[0][0], UU, 32 * k4, lane), whr[k4], r);
#pragma unroll
    for (int i = 0; i < 8; ++i) r[i] = sigmoidf(r[i]);

    // z = sigmoid(xz + h_d @ W_hz)
    v8f z;
#pragma unroll
    for (int i = 0; i < 8; ++i) z[i] = (float)st[2][i + hi8][col];
#pragma unroll
    for (int k4 = 0; k4 < 4; ++k4)
      z = wmma_step(load_a_frag(&sHd[0][0], UU, 32 * k4, lane), whz[k4], z);
#pragma unroll
    for (int i = 0; i < 8; ++i) z[i] = sigmoidf(z[i]);

    // publish r*h_d to LDS
#pragma unroll
    for (int i = 0; i < 8; ++i) sRh[i + hi8][col] = (_Float16)(r[i] * hd[i]);
    __syncthreads();

    // h_base = tanh(xh + (r*h_d) @ W_hh)
    v8f hb;
#pragma unroll
    for (int i = 0; i < 8; ++i) hb[i] = (float)st[3][i + hi8][col];
#pragma unroll
    for (int k4 = 0; k4 < 4; ++k4)
      hb = wmma_step(load_a_frag(&sRh[0][0], UU, 32 * k4, lane), whh[k4], hb);

#pragma unroll
    for (int i = 0; i < 8; ++i) {
      const float hbase = tanhf(hb[i]);
      const float hfreq = tanhf(hbase + (float)st[4][i + hi8][col]);
      const float ps    = (float)st[5][i + hi8][col];
      const float hcand = (1.0f - ps) * hbase + ps * hfreq;
      // alpha == 1 (softmax over singleton axis) -> h_att = h_d
      h[i] = (1.0f - z[i]) * hd[i] + z[i] * hcand;
    }
    __syncthreads();  // reads of sHd/sRh/sT[buf] done before reuse
    buf ^= 1;
  }

  // write final hidden state (f16, feeds classifier GEMM A operand)
#pragma unroll
  for (int i = 0; i < 8; ++i)
    g_h16[(size_t)(rb0 + i + hi8) * UU + col] = (_Float16)h[i];
}

// ---------------- Kernel 3: classifier logits -------------------------------
// logits[m][n] = e_long @ W_out + b_out ; 16x16 WMMA tiles, 50000 = 3125*16.
__global__ __launch_bounds__(256) void out_kernel(
    const _Float16* __restrict__ g_h16, const _Float16* __restrict__ WtOut,
    const float* __restrict__ b_out, float* __restrict__ logits) {
  const int tid  = threadIdx.x;
  const int lane = tid & 31;
  const int w    = tid >> 5;
  const int nt   = blockIdx.x * 8 + w;     // N tile (wave-uniform)
  if (nt >= CC / 16) return;
  const int mt   = blockIdx.y;             // M tile
  const int nloc = lane & 15;
  const int hi8  = (lane < 16) ? 0 : 8;
  const int col  = nt * 16 + nloc;

  // pull next tile's weight block toward the caches
  if (nt + 8 < CC / 16)
    __builtin_prefetch(WtOut + (size_t)(nt + 8) * 16 * UU, 0, 1);

  v8f acc; const float bb = b_out[col];
#pragma unroll
  for (int i = 0; i < 8; ++i) acc[i] = bb;

  const _Float16* aBase = g_h16 + (size_t)mt * 16 * UU;
#pragma unroll
  for (int k4 = 0; k4 < 4; ++k4)
    acc = wmma_step(load_a_frag(aBase, UU, 32 * k4, lane),
                    load_b_frag(WtOut, UU, nt * 16, 32 * k4, lane), acc);

#pragma unroll
  for (int i = 0; i < 8; ++i)
    logits[(size_t)(mt * 16 + i + hi8) * CC + col] = acc[i];
}

// ---------------- Kernel 4: row softmax over 50000 classes ------------------
__global__ __launch_bounds__(256) void softmax_kernel(float* __restrict__ x) {
  __shared__ float red[256];
  const int tid = threadIdx.x;
  const size_t base = (size_t)blockIdx.x * CC;

  float m = -3.402823466e38f;
  for (int i = tid; i < CC; i += 256) m = fmaxf(m, x[base + i]);
  red[tid] = m; __syncthreads();
  for (int s = 128; s > 0; s >>= 1) {
    if (tid < s) red[tid] = fmaxf(red[tid], red[tid + s]);
    __syncthreads();
  }
  const float mx = red[0]; __syncthreads();

  float sum = 0.0f;
  for (int i = tid; i < CC; i += 256) sum += __expf(x[base + i] - mx);
  red[tid] = sum; __syncthreads();
  for (int s = 128; s > 0; s >>= 1) {
    if (tid < s) red[tid] += red[tid + s];
    __syncthreads();
  }
  const float inv = 1.0f / red[0]; __syncthreads();

  for (int i = tid; i < CC; i += 256)
    x[base + i] = __expf(x[base + i] - mx) * inv;
}

// ---------------------------------------------------------------------------

extern "C" void kernel_launch(void* const* d_in, const int* in_sizes, int n_in,
                              void* d_out, int out_size, void* d_ws,
                              size_t ws_size, hipStream_t stream) {
  (void)in_sizes; (void)n_in; (void)out_size; (void)ws_size;

  const int*   item   = (const int*)d_in[0];
  const int*   tix    = (const int*)d_in[1];
  const int*   frq    = (const int*)d_in[2];
  const float* emb_i  = (const float*)d_in[3];
  const float* emb_t  = (const float*)d_in[4];
  const float* emb_f  = (const float*)d_in[5];
  const float* W_xr   = (const float*)d_in[6];
  const float* W_hr   = (const float*)d_in[7];
  const float* b_r    = (const float*)d_in[8];
  const float* W_xz   = (const float*)d_in[9];
  const float* W_hz   = (const float*)d_in[10];
  const float* b_z    = (const float*)d_in[11];
  const float* W_xh   = (const float*)d_in[12];
  const float* W_hh   = (const float*)d_in[13];
  const float* b_h    = (const float*)d_in[14];
  const float* W_xtg  = (const float*)d_in[15];
  const float* W_tg   = (const float*)d_in[16];
  const float* b_tg   = (const float*)d_in[17];
  const float* W_xfg  = (const float*)d_in[18];
  const float* W_fg   = (const float*)d_in[19];
  const float* b_fg   = (const float*)d_in[20];
  const float* W_delta= (const float*)d_in[21];
  const float* b_delta= (const float*)d_in[22];
  const float* W_fdir = (const float*)d_in[23];
  const float* b_fdir = (const float*)d_in[24];
  const float* W_psi  = (const float*)d_in[25];
  const float* b_psi  = (const float*)d_in[26];
  // d_in[27] = W_a: unused (softmax over singleton axis -> alpha == 1)
  const float* W_out  = (const float*)d_in[28];
  const float* b_out  = (const float*)d_in[29];

  float* logits = (float*)d_out;

  // ---- carve workspace (≈92 MB) ----
  char* ws = (char*)d_ws;
  size_t off = 0;
  auto carve = [&](size_t bytes) -> char* {
    char* p = ws + off;
    off += (bytes + 255) & ~(size_t)255;
    return p;
  };
  const size_t WSM = (size_t)UU * UU * sizeof(_Float16);   // 32 KB
  _Float16* WtXr   = (_Float16*)carve(WSM);
  _Float16* WtXz   = (_Float16*)carve(WSM);
  _Float16* WtXh   = (_Float16*)carve(WSM);
  _Float16* WtXtg  = (_Float16*)carve(WSM);
  _Float16* WtTg   = (_Float16*)carve(WSM);
  _Float16* WtXfg  = (_Float16*)carve(WSM);
  _Float16* WtFg   = (_Float16*)carve(WSM);
  _Float16* WtFdir = (_Float16*)carve(WSM);
  _Float16* WtPsi  = (_Float16*)carve(WSM);
  _Float16* WtHr   = (_Float16*)carve(WSM);
  _Float16* WtHz   = (_Float16*)carve(WSM);
  _Float16* WtHh   = (_Float16*)carve(WSM);
  _Float16* WtDelta= (_Float16*)carve((size_t)2 * UU * UU * sizeof(_Float16));
  _Float16* WtOut  = (_Float16*)carve((size_t)CC * UU * sizeof(_Float16));
  const size_t ACT = (size_t)MM * UU * sizeof(_Float16);   // 13.1 MB
  _Float16* g_xr    = (_Float16*)carve(ACT);
  _Float16* g_xz    = (_Float16*)carve(ACT);
  _Float16* g_xh    = (_Float16*)carve(ACT);
  _Float16* g_decay = (_Float16*)carve(ACT);
  _Float16* g_fdir  = (_Float16*)carve(ACT);
  _Float16* g_psi   = (_Float16*)carve(ACT);
  _Float16* g_h16   = (_Float16*)carve((size_t)BB * UU * sizeof(_Float16));

  // ---- 0) weight convert/transpose to f16 Wt[N][K] ----
  auto wt = [&](const float* src, _Float16* dst, int K, int N) {
    size_t total = (size_t)K * N;
    int grid = (int)((total + 255) / 256);
    if (grid > 8192) grid = 8192;
    wtrans_kernel<<<grid, 256, 0, stream>>>(src, dst, K, N);
  };
  wt(W_xr,  WtXr,  DD, UU);  wt(W_xz,  WtXz,  DD, UU);  wt(W_xh,  WtXh,  DD, UU);
  wt(W_xtg, WtXtg, DD, UU);  wt(W_tg,  WtTg,  DD, UU);
  wt(W_xfg, WtXfg, DD, UU);  wt(W_fg,  WtFg,  DD, UU);
  wt(W_fdir,WtFdir, UU, UU); wt(W_psi, WtPsi, UU, UU);
  wt(W_hr,  WtHr,  UU, UU);  wt(W_hz,  WtHz,  UU, UU);  wt(W_hh,  WtHh,  UU, UU);
  wt(W_delta, WtDelta, 2 * UU, UU);
  wt(W_out,   WtOut,   UU, CC);

  // ---- 1) fused gather + all per-timestep projections (WMMA) ----
  proj_kernel<<<MM / 16, 256, 0, stream>>>(
      item, tix, frq, emb_i, emb_t, emb_f,
      WtXr, WtXz, WtXh, WtXtg, WtTg, WtXfg, WtFg, WtDelta, WtFdir, WtPsi,
      b_r, b_z, b_h, b_tg, b_fg, b_delta, b_fdir, b_psi,
      g_xr, g_xz, g_xh, g_decay, g_fdir, g_psi);

  // ---- 2) recurrent scan over S=200 steps (WMMA + async double-buffer) ----
  scan_kernel<<<BB / 16, 256, 0, stream>>>(
      WtHr, WtHz, WtHh, g_xr, g_xz, g_xh, g_decay, g_fdir, g_psi, g_h16);

  // ---- 3) classifier logits (WMMA) ----
  dim3 ogrid((CC / 16 + 7) / 8, BB / 16);
  out_kernel<<<ogrid, 256, 0, stream>>>(g_h16, WtOut, b_out, logits);

  // ---- 4) row softmax ----
  softmax_kernel<<<BB, 256, 0, stream>>>(logits);
}
